// GCN_multi_feature_8229157339650
// MI455X (gfx1250) — matrix operand order repaired and checked by reading, not compiled, not run on hardware
//
#include <hip/hip_runtime.h>
#include <math.h>

// ---------------------------------------------------------------------------
// CDNA5 (gfx1250) implementation: wave32, WMMA f32_16x16x32_f16 GEMM core,
// software-pipelined global->reg->LDS staging.
// ---------------------------------------------------------------------------

typedef _Float16 f16;
typedef f16  v16h __attribute__((ext_vector_type(16)));
typedef f16  v8h  __attribute__((ext_vector_type(8)));
typedef float v8f __attribute__((ext_vector_type(8)));

#define EPI_NONE      0
#define EPI_BIAS_RELU 1
#define EPI_SIM       2

static constexpr int NN   = 4096;       // nodes
static constexpr int KNN  = 30;         // neighbors
static constexpr int EPR  = KNN + 1;    // knn edges per row incl. self
static constexpr int TS   = 40;         // LDS tile row stride (f16): 80B, 16B aligned

// ---------------------------------------------------------------------------
// Tiled WMMA GEMM:  C[M x N] = A[M x K] * B  (+ epilogue)
//   BT=true : B is a weight matrix stored [K x N] (row-major), transposed on
//             the way into LDS so fragments load contiguously.
//   BT=false: B is row-major [N x K] (used for sim = H * H^T).
// Block = 128x128 tile, 8 waves (2 x 4), each wave: 4x2 tiles of 16x16.
// Pipelined: tile k+1 is fetched into registers while tile k is computed.
// ---------------------------------------------------------------------------
template <int EPI, bool BT>
__global__ __launch_bounds__(256) void gemm_kernel(
    const float* __restrict__ Ag, int lda,
    const float* __restrict__ Bg, int ldb,
    float* __restrict__ Cg, int ldc, int Kdim,
    const float* __restrict__ bias,   // EPI_BIAS_RELU
    const float* __restrict__ rn)     // EPI_SIM: reciprocal row norms
{
  __shared__ __align__(16) f16 lA[128 * TS];
  __shared__ __align__(16) f16 lB[128 * TS];

  const int t    = threadIdx.x;
  const int lane = t & 31;
  const int wave = t >> 5;
  const int wm   = wave >> 2;      // 0..1
  const int wn   = wave & 3;       // 0..3
  const int l15  = lane & 15;
  const int kh   = lane >> 4;      // 0..1
  const int m0   = blockIdx.y * 128;
  const int n0   = blockIdx.x * 128;

  // per-thread staging coordinates
  const int arow = t >> 1, ak0 = (t & 1) * 16;  // A (and B when !BT)
  const int bkq = t >> 7, bn = t & 127;         // B when BT

  v8f acc[4][2];
#pragma unroll
  for (int m = 0; m < 4; ++m)
#pragma unroll
    for (int n = 0; n < 2; ++n)
#pragma unroll
      for (int i = 0; i < 8; ++i) acc[m][n][i] = 0.0f;

  float4 ra[4];   // staged A slice (16 fp32)
  float4 rb4[4];  // staged B slice (!BT)
  float  rbs[16]; // staged B slice (BT)

  auto fetch = [&](int kb) {
    const float4* ap = (const float4*)(Ag + (size_t)(m0 + arow) * lda + kb + ak0);
#pragma unroll
    for (int q = 0; q < 4; ++q) ra[q] = ap[q];
    if (BT) {
#pragma unroll
      for (int kk = 0; kk < 16; ++kk)
        rbs[kk] = Bg[(size_t)(kb + bkq * 16 + kk) * ldb + n0 + bn];
    } else {
      const float4* bp = (const float4*)(Bg + (size_t)(n0 + arow) * ldb + kb + ak0);
#pragma unroll
      for (int q = 0; q < 4; ++q) rb4[q] = bp[q];
    }
  };
  auto stage = [&]() {
#pragma unroll
    for (int q = 0; q < 4; ++q) {
      int o = arow * TS + ak0 + q * 4;
      lA[o + 0] = (f16)ra[q].x; lA[o + 1] = (f16)ra[q].y;
      lA[o + 2] = (f16)ra[q].z; lA[o + 3] = (f16)ra[q].w;
    }
    if (BT) {
#pragma unroll
      for (int kk = 0; kk < 16; ++kk)
        lB[bn * TS + bkq * 16 + kk] = (f16)rbs[kk];
    } else {
#pragma unroll
      for (int q = 0; q < 4; ++q) {
        int o = arow * TS + ak0 + q * 4;
        lB[o + 0] = (f16)rb4[q].x; lB[o + 1] = (f16)rb4[q].y;
        lB[o + 2] = (f16)rb4[q].z; lB[o + 3] = (f16)rb4[q].w;
      }
    }
  };

  fetch(0);
  for (int kb = 0; kb < Kdim; kb += 32) {
    stage();
    __syncthreads();
    if (kb + 32 < Kdim) fetch(kb + 32);  // in flight during WMMA

    // ---- fragments per ISA 16-bit layouts ----
    v16h afr[4], bfr[2];
#pragma unroll
    for (int m = 0; m < 4; ++m) {
      int r = wm * 64 + m * 16 + l15;
      v8h lo = *(const v8h*)&lA[r * TS + kh * 8];        // K 0..7 / 8..15
      v8h hi = *(const v8h*)&lA[r * TS + 16 + kh * 8];   // K 16..23 / 24..31
#pragma unroll
      for (int i = 0; i < 8; ++i) { afr[m][i] = lo[i]; afr[m][i + 8] = hi[i]; }
    }
#pragma unroll
    for (int n = 0; n < 2; ++n) {
      int r = wn * 32 + n * 16 + l15;
      v8h lo = *(const v8h*)&lB[r * TS + kh * 8];        // K 0..7 / 8..15
      v8h hi = *(const v8h*)&lB[r * TS + 16 + kh * 8];   // K 16..23 / 24..31
#pragma unroll
      for (int i = 0; i < 8; ++i) { bfr[n][i] = lo[i]; bfr[n][i + 8] = hi[i]; }
    }
#pragma unroll
    for (int m = 0; m < 4; ++m)
#pragma unroll
      for (int n = 0; n < 2; ++n)
        acc[m][n] = __builtin_amdgcn_wmma_f32_16x16x32_f16(
            false, afr[m], false, bfr[n], (short)0, acc[m][n], false, false);
    __syncthreads();
  }

  // ---- epilogue / store (C layout: VGPR v -> M = v + kh*8, N = l15) ----
#pragma unroll
  for (int m = 0; m < 4; ++m) {
    int rbase = m0 + wm * 64 + m * 16 + kh * 8;
#pragma unroll
    for (int n = 0; n < 2; ++n) {
      int col = n0 + wn * 32 + n * 16 + l15;
      float bv = (EPI == EPI_BIAS_RELU) ? bias[col] : 0.0f;
      float rc = (EPI == EPI_SIM) ? rn[col] : 0.0f;
      float* cp = Cg + (size_t)rbase * ldc + col;
      const float* rp = (EPI == EPI_SIM) ? (rn + rbase) : nullptr;
#pragma unroll
      for (int v = 0; v < 8; ++v) {
        float x = acc[m][n][v];
        if (EPI == EPI_BIAS_RELU) x = fmaxf(x + bv, 0.0f);
        else if (EPI == EPI_SIM) {
          x *= rc * rp[v];
          if (rbase + v == col) x = -__builtin_inff();
        }
        *cp = x;
        cp += ldc;
      }
    }
  }
}

// reciprocal L2 row norms (one wave per row, 8 rows per block)
__global__ __launch_bounds__(256) void rownorm_kernel(
    const float* __restrict__ h, int ld, int Kd, float* __restrict__ rn) {
  int wave = threadIdx.x >> 5, lane = threadIdx.x & 31;
  int n = blockIdx.x * 8 + wave;
  const float* p = h + (size_t)n * ld;
  float s = 0.0f;
  for (int i = lane; i < Kd; i += 32) { float v = p[i]; s += v * v; }
#pragma unroll
  for (int m = 16; m; m >>= 1) s += __shfl_xor(s, m, 32);
  if (lane == 0) rn[n] = 1.0f / fmaxf(sqrtf(s), 1e-12f);
}

// top-30 per similarity row -> CSR (31 entries/row incl. self, w = 1/31)
__global__ __launch_bounds__(256) void topk_kernel(
    const float* __restrict__ sim, int* __restrict__ col,
    float* __restrict__ w) {
  __shared__ float srow[NN];
  __shared__ float smax[256];
  __shared__ int   sidx[256];
  const int row = blockIdx.x, t = threadIdx.x;
  for (int i = t; i < NN; i += 256) srow[i] = sim[(size_t)row * NN + i];
  __syncthreads();
  for (int it = 0; it < KNN; ++it) {
    float bv = -3.0e38f; int bi = 0;
    for (int i = t; i < NN; i += 256) {
      float v = srow[i];
      if (v > bv) { bv = v; bi = i; }
    }
    smax[t] = bv; sidx[t] = bi;
    __syncthreads();
    for (int s = 128; s > 0; s >>= 1) {
      if (t < s && smax[t + s] > smax[t]) { smax[t] = smax[t + s]; sidx[t] = sidx[t + s]; }
      __syncthreads();
    }
    if (t == 0) {
      col[row * EPR + it] = sidx[0];
      w[row * EPR + it]   = 1.0f / (float)EPR;
      srow[sidx[0]] = -3.0e38f;
    }
    __syncthreads();
  }
  if (t == 0) { col[row * EPR + KNN] = row; w[row * EPR + KNN] = 1.0f / (float)EPR; }
}

// CSR gather: out[n,f] = sum_e w[e]*hin[col[e],f] + bias[f] (+ resid[n,f])
__global__ __launch_bounds__(256) void gather_kernel(
    const int* __restrict__ colidx, const float* __restrict__ wgt,
    const int* __restrict__ row_ptr, int fixed_per_row,
    const float* __restrict__ hin, int ldin,
    const float* __restrict__ bias,
    const float* __restrict__ resid, int ldres,
    float* __restrict__ out, int ldout, int F) {
  const int n = blockIdx.x, t = threadIdx.x;
  const int s0 = row_ptr ? row_ptr[n]     : n * fixed_per_row;
  const int s1 = row_ptr ? row_ptr[n + 1] : s0 + fixed_per_row;
  const int nf = F >> 8;  // F in {256,512,1024}
  float acc[4] = {0.f, 0.f, 0.f, 0.f};
  for (int e = s0; e < s1; ++e) {
    int c = colidx[e];
    float wv = wgt[e];
    const float* hp = hin + (size_t)c * ldin;
#pragma unroll
    for (int j = 0; j < 4; ++j)
      if (j < nf) acc[j] += wv * hp[t + (j << 8)];
  }
#pragma unroll
  for (int j = 0; j < 4; ++j)
    if (j < nf) {
      int f = t + (j << 8);
      float v = acc[j] + bias[f];
      if (resid) v += resid[(size_t)n * ldres + f];
      out[(size_t)n * ldout + f] = v;
    }
}

// ---- random-graph CSR construction ----
__global__ void fill_int_kernel(int* p, int v, int n) {
  int i = blockIdx.x * 256 + threadIdx.x;
  if (i < n) p[i] = v;
}
__global__ void deg_kernel(const int* __restrict__ dst, int E, int* cnt) {
  int e = blockIdx.x * 256 + threadIdx.x;
  if (e < E) atomicAdd(&cnt[dst[e]], 1);
}
__global__ __launch_bounds__(256) void prefix_kernel(
    const int* __restrict__ cnt, int* row_ptr, int* cursor, float* dinv) {
  for (int i = threadIdx.x; i < NN; i += 256)
    dinv[i] = rsqrtf((float)cnt[i]);
  if (threadIdx.x == 0) {
    int a = 0;
    for (int i = 0; i < NN; ++i) { row_ptr[i] = a; cursor[i] = a; a += cnt[i]; }
    row_ptr[NN] = a;
  }
}
__global__ void fill_edges_kernel(const int* __restrict__ src,
                                  const int* __restrict__ dst, int E,
                                  const float* __restrict__ dinv, int* cursor,
                                  int* colx, float* wx) {
  int e = blockIdx.x * 256 + threadIdx.x;
  if (e < E) {
    int s = src[e], d = dst[e];
    int pos = atomicAdd(&cursor[d], 1);
    colx[pos] = s;
    wx[pos] = dinv[s] * dinv[d];
  } else if (e < E + NN) {
    int n_ = e - E;
    int pos = atomicAdd(&cursor[n_], 1);
    colx[pos] = n_;
    float di = dinv[n_];
    wx[pos] = di * di;
  }
}

// head: out[n] = sigmoid(dot(h[n,0:128], W) + b)
__global__ __launch_bounds__(256) void fc2_kernel(
    const float* __restrict__ hin, const float* __restrict__ W,
    const float* __restrict__ b, float* __restrict__ out) {
  int wave = threadIdx.x >> 5, lane = threadIdx.x & 31;
  int n = blockIdx.x * 8 + wave;
  const float* p = hin + (size_t)n * 128;
  float s = 0.0f;
  for (int i = lane; i < 128; i += 32) s += p[i] * W[i];
#pragma unroll
  for (int m = 16; m; m >>= 1) s += __shfl_xor(s, m, 32);
  if (lane == 0) out[n] = 1.0f / (1.0f + expf(-(s + b[0])));
}

// ---------------------------------------------------------------------------
// Workspace layout
// ---------------------------------------------------------------------------
static constexpr size_t alup(size_t x) { return (x + 255) & ~(size_t)255; }
static constexpr size_t OFF_CONCAT = 0;
static constexpr size_t SZ_CONCAT  = alup((size_t)NN * 3328 * 4);
static constexpr size_t OFF_H      = OFF_CONCAT + SZ_CONCAT;
static constexpr size_t SZ_H       = alup((size_t)NN * 3328 * 4);
static constexpr size_t OFF_T      = OFF_H + SZ_H;
static constexpr size_t SZ_T       = alup((size_t)NN * 1024 * 4);
static constexpr size_t OFF_SIM    = OFF_T + SZ_T;
static constexpr size_t SZ_SIM     = alup((size_t)NN * NN * 4);
static constexpr size_t OFF_RN     = OFF_SIM + SZ_SIM;
static constexpr size_t OFF_DINV   = OFF_RN + alup(NN * 4);
static constexpr size_t OFF_CNT    = OFF_DINV + alup(NN * 4);
static constexpr size_t OFF_RP     = OFF_CNT + alup(NN * 4);
static constexpr size_t OFF_CUR    = OFF_RP + alup((NN + 1) * 4);
static constexpr size_t OFF_COL    = OFF_CUR + alup(NN * 4);
static constexpr size_t OFF_W      = OFF_COL + alup((size_t)NN * EPR * 4);
static constexpr size_t WS_NEEDED  = OFF_W + alup((size_t)NN * EPR * 4);

extern "C" void kernel_launch(void* const* d_in, const int* in_sizes, int n_in,
                              void* d_out, int out_size, void* d_ws,
                              size_t ws_size, hipStream_t stream) {
  if (ws_size < WS_NEEDED) return;  // cannot run without scratch

  char* ws = (char*)d_ws;
  float* CONCAT = (float*)(ws + OFF_CONCAT);
  float* HB     = (float*)(ws + OFF_H);
  float* TB     = (float*)(ws + OFF_T);
  float* SIM    = (float*)(ws + OFF_SIM);
  float* RN     = (float*)(ws + OFF_RN);
  float* DINV   = (float*)(ws + OFF_DINV);
  int*   CNT    = (int*)(ws + OFF_CNT);
  int*   RP     = (int*)(ws + OFF_RP);
  int*   CUR    = (int*)(ws + OFF_CUR);
  int*   COL    = (int*)(ws + OFF_COL);
  float* WGT    = (float*)(ws + OFF_W);
  float* OUT    = (float*)d_out;

  auto P = [&](int i) { return (const float*)d_in[10 + i]; };
  const int E = in_sizes[5] / 2;  // edges per input graph (N*30)

  struct Br {
    const float* x; const int* ei; int d; int coloff;
    const float *W1, *b1, *W2, *b2;
  };
  Br br[5] = {
      {(const float*)d_in[0], (const int*)d_in[5], 1024, 0,    P(0),  P(1),  P(2),  P(3)},
      {(const float*)d_in[1], (const int*)d_in[6], 1024, 1024, P(4),  P(5),  P(6),  P(7)},
      {(const float*)d_in[2], (const int*)d_in[7], 512,  2048, P(8),  P(9),  P(10), P(11)},
      {(const float*)d_in[3], (const int*)d_in[8], 512,  2560, P(12), P(13), P(14), P(15)},
      {(const float*)d_in[4], (const int*)d_in[9], 256,  3072, P(16), P(17), P(18), P(19)},
  };

  const dim3 blk(256);
  auto simgrid = dim3(NN / 128, NN / 128);

  // ---------------- per-branch pipeline ----------------
  for (int b = 0; b < 5; ++b) {
    const int d = br[b].d;
    dim3 g(d / 128, NN / 128);

    // t = x @ W1
    gemm_kernel<EPI_NONE, true><<<g, blk, 0, stream>>>(
        br[b].x, d, br[b].W1, d, TB, d, d, nullptr, nullptr);

    // CSR of the input random graph (dst = ei + E), incl. self loops
    fill_int_kernel<<<(NN + 255) / 256, blk, 0, stream>>>(CNT, 1, NN);
    deg_kernel<<<(E + 255) / 256, blk, 0, stream>>>(br[b].ei + E, E, CNT);
    prefix_kernel<<<1, blk, 0, stream>>>(CNT, RP, CUR, DINV);
    fill_edges_kernel<<<(E + NN + 255) / 256, blk, 0, stream>>>(
        br[b].ei, br[b].ei + E, E, DINV, CUR, COL, WGT);

    // h = gather(t) + b1 + x
    gather_kernel<<<NN, blk, 0, stream>>>(COL, WGT, RP, 0, TB, d, br[b].b1,
                                          br[b].x, d, HB, d, d);
    // cosine kNN on h
    rownorm_kernel<<<NN / 8, blk, 0, stream>>>(HB, d, d, RN);
    gemm_kernel<EPI_SIM, false><<<simgrid, blk, 0, stream>>>(
        HB, d, HB, d, SIM, NN, d, nullptr, RN);
    topk_kernel<<<NN, blk, 0, stream>>>(SIM, COL, WGT);
    // t2 = h @ W2 ; gather into concat slice
    gemm_kernel<EPI_NONE, true><<<g, blk, 0, stream>>>(
        HB, d, br[b].W2, d, TB, d, d, nullptr, nullptr);
    gather_kernel<<<NN, blk, 0, stream>>>(COL, WGT, nullptr, EPR, TB, d,
                                          br[b].b2, nullptr, 0,
                                          CONCAT + br[b].coloff, 3328, d);
  }

  // ---------------- trunk ----------------
  // A = relu(concat @ a_W + a_b)   [4096 x 3328]
  gemm_kernel<EPI_BIAS_RELU, true><<<dim3(3328 / 128, NN / 128), blk, 0, stream>>>(
      CONCAT, 3328, P(20), 3328, HB, 3328, 3328, P(21), nullptr);
  // B2 = relu(A @ b_W + b_b)       [4096 x 1024]
  gemm_kernel<EPI_BIAS_RELU, true><<<dim3(1024 / 128, NN / 128), blk, 0, stream>>>(
      HB, 3328, P(22), 1024, TB, 1024, 3328, P(23), nullptr);

  // kNN(B2); g4: gemm -> gather    [-> 512]
  rownorm_kernel<<<NN / 8, blk, 0, stream>>>(TB, 1024, 1024, RN);
  gemm_kernel<EPI_SIM, false><<<simgrid, blk, 0, stream>>>(
      TB, 1024, TB, 1024, SIM, NN, 1024, nullptr, RN);
  topk_kernel<<<NN, blk, 0, stream>>>(SIM, COL, WGT);
  gemm_kernel<EPI_NONE, true><<<dim3(512 / 128, NN / 128), blk, 0, stream>>>(
      TB, 1024, P(24), 512, CONCAT, 512, 1024, nullptr, nullptr);
  gather_kernel<<<NN, blk, 0, stream>>>(COL, WGT, nullptr, EPR, CONCAT, 512,
                                        P(25), nullptr, 0, TB, 512, 512);

  // kNN(h4); g5: gemm -> gather    [-> 256]
  rownorm_kernel<<<NN / 8, blk, 0, stream>>>(TB, 512, 512, RN);
  gemm_kernel<EPI_SIM, false><<<simgrid, blk, 0, stream>>>(
      TB, 512, TB, 512, SIM, NN, 512, nullptr, RN);
  topk_kernel<<<NN, blk, 0, stream>>>(SIM, COL, WGT);
  gemm_kernel<EPI_NONE, true><<<dim3(256 / 128, NN / 128), blk, 0, stream>>>(
      TB, 512, P(26), 256, CONCAT, 256, 512, nullptr, nullptr);
  gather_kernel<<<NN, blk, 0, stream>>>(COL, WGT, nullptr, EPR, CONCAT, 256,
                                        P(27), nullptr, 0, HB, 256, 256);

  // fc1 relu -> fc2 sigmoid
  gemm_kernel<EPI_BIAS_RELU, true><<<dim3(1, NN / 128), blk, 0, stream>>>(
      HB, 256, P(28), 128, TB, 128, 256, P(29), nullptr);
  fc2_kernel<<<NN / 8, blk, 0, stream>>>(TB, P(30), P(31), OUT);

  (void)n_in; (void)out_size;
}
